// DiscreteDynamicHyperNetSIR_41566693490891
// MI455X (gfx1250) — matrix operand order repaired and checked by reading, not compiled, not run on hardware
//
#include <hip/hip_runtime.h>
#include <hip/hip_bf16.h>
#include <math.h>

typedef __attribute__((ext_vector_type(16))) _Float16 v16h;
typedef __attribute__((ext_vector_type(8)))  float    v8f;

#define WH_PAD 1024   // padded edge count for K=32 WMMA chunks (E=1000 <= 1024)

// ---------------------------------------------------------------------------
// Init: out[0] = x, states = argmax(x), vh = (state==I) in f16
// ---------------------------------------------------------------------------
__global__ void k_init(const float* __restrict__ x, float* __restrict__ out0,
                       int* __restrict__ states, _Float16* __restrict__ vh, int N) {
  int n = blockIdx.x * blockDim.x + threadIdx.x;
  if (n >= N) return;
  float x0 = x[3 * n + 0], x1 = x[3 * n + 1], x2 = x[3 * n + 2];
  out0[3 * n + 0] = x0; out0[3 * n + 1] = x1; out0[3 * n + 2] = x2;
  int s = 0; float m = x0;
  if (x1 > m) { m = x1; s = 1; }
  if (x2 > m) { s = 2; }
  states[n] = s;
  vh[n] = (s == 1) ? (_Float16)1.0f : (_Float16)0.0f;
}

// ---------------------------------------------------------------------------
// Kernel 1: c_partial = (H_t @ v) split-K via WMMA f16.
// One wave per block. A = 16 edges x 32 nodes of H (cvt f32->f16),
// B = v chunk in column N=0 only (lanes 0 and 16 carry K=0..15 / 16..31).
// D column 0 lands in lane0 (M=0..7) and lane16 (M=8..15), VGPRs 0..7.
// ---------------------------------------------------------------------------
__global__ void k_edge(const float* __restrict__ Hstep, const _Float16* __restrict__ vh,
                       float* __restrict__ cpart, int N, int chunksN, int cps) {
  const int lane   = threadIdx.x;
  const int laneLo = lane & 15;
  const int laneHi = lane >> 4;
  const size_t e   = (size_t)blockIdx.x * 16 + laneLo;   // may over-read rows >= E: valid {0,1} memory
  const float* rowp = Hstep + e * (size_t)N;

  v8f acc;
  #pragma unroll
  for (int i = 0; i < 8; ++i) acc[i] = 0.0f;

  int c0 = blockIdx.y * cps;
  int c1 = c0 + cps; if (c1 > chunksN) c1 = chunksN;

  for (int c = c0; c < c1; ++c) {
    const int nbase = c * 32;
    const float* rp = rowp + nbase + (laneHi << 4);
    float4 f0 = *(const float4*)(rp + 0);
    float4 f1 = *(const float4*)(rp + 4);
    float4 f2 = *(const float4*)(rp + 8);
    float4 f3 = *(const float4*)(rp + 12);
    v16h a;
    a[0]=(_Float16)f0.x;  a[1]=(_Float16)f0.y;  a[2]=(_Float16)f0.z;  a[3]=(_Float16)f0.w;
    a[4]=(_Float16)f1.x;  a[5]=(_Float16)f1.y;  a[6]=(_Float16)f1.z;  a[7]=(_Float16)f1.w;
    a[8]=(_Float16)f2.x;  a[9]=(_Float16)f2.y;  a[10]=(_Float16)f2.z; a[11]=(_Float16)f2.w;
    a[12]=(_Float16)f3.x; a[13]=(_Float16)f3.y; a[14]=(_Float16)f3.z; a[15]=(_Float16)f3.w;

    v16h b;
    #pragma unroll
    for (int h = 0; h < 16; ++h) b[h] = (_Float16)0.0f;
    if (laneLo == 0) b = *(const v16h*)(vh + nbase + laneHi * 16);

    acc = __builtin_amdgcn_wmma_f32_16x16x32_f16(false, a, false, b, (short)0, acc,
                                                 false, false);
  }

  if (laneLo == 0) {
    float* dst = cpart + (size_t)blockIdx.y * WH_PAD + (size_t)blockIdx.x * 16 + laneHi * 8;
    #pragma unroll
    for (int r = 0; r < 8; ++r) dst[r] = acc[r];
  }
}

// ---------------------------------------------------------------------------
// Reduce split-K partials, w = tanh(c), convert to f16; pad e>=E with 0.
// ---------------------------------------------------------------------------
__global__ void k_tanh(const float* __restrict__ cpart, int nsplits,
                       _Float16* __restrict__ wh, int E) {
  for (int e = threadIdx.x; e < WH_PAD; e += blockDim.x) {
    if (e < E) {
      float s = 0.0f;
      for (int k = 0; k < nsplits; ++k) s += cpart[(size_t)k * WH_PAD + e];
      wh[e] = (_Float16)tanhf(s);
    } else {
      wh[e] = (_Float16)0.0f;
    }
  }
}

// ---------------------------------------------------------------------------
// Kernel 2: agg_partial = w^T @ H_t split over edges via WMMA f16.
// One wave per block. A = w in row M=0 only (lanes 0/16), B = 32 edges x 16
// nodes of H. D row M=0 = lanes 0..15, VGPR0. Rows e>=E contribute 0 via
// wh padding (H over-read values are finite {0,1}: no 0*inf).
// ---------------------------------------------------------------------------
__global__ void k_agg(const float* __restrict__ Hstep, const _Float16* __restrict__ wh,
                      float* __restrict__ aggpart, int N, int chunksE, int cps) {
  const int lane   = threadIdx.x;
  const int laneLo = lane & 15;
  const int laneHi = lane >> 4;
  const int nbase  = blockIdx.x * 16;

  v8f acc;
  #pragma unroll
  for (int i = 0; i < 8; ++i) acc[i] = 0.0f;

  int c0 = blockIdx.y * cps;
  int c1 = c0 + cps; if (c1 > chunksE) c1 = chunksE;

  for (int c = c0; c < c1; ++c) {
    const int ebase = c * 32;
    // B: lane holds H[ebase + laneHi*16 + h][nbase + laneLo], h = 0..15
    const float* col = Hstep + (size_t)(ebase + laneHi * 16) * N + nbase + laneLo;
    v16h b;
    #pragma unroll
    for (int h = 0; h < 16; ++h) b[h] = (_Float16)col[(size_t)h * N];

    // A: row M=0 = w; lanes with laneLo==0 carry it, everyone else zero
    v16h a;
    #pragma unroll
    for (int h = 0; h < 16; ++h) a[h] = (_Float16)0.0f;
    if (laneLo == 0) {
      const int eh = ebase + laneHi * 16;
      #pragma unroll
      for (int h = 0; h < 8; ++h) a[h] = wh[eh + h];
      #pragma unroll
      for (int h = 0; h < 8; ++h) a[8 + h] = wh[eh + 8 + h];
    }

    acc = __builtin_amdgcn_wmma_f32_16x16x32_f16(false, a, false, b, (short)0, acc,
                                                 false, false);
  }

  if (laneHi == 0)
    aggpart[(size_t)blockIdx.y * N + nbase + laneLo] = acc[0];
}

// ---------------------------------------------------------------------------
// Per-node SIR update, replicating the reference arithmetic order in f32.
// ---------------------------------------------------------------------------
__global__ void k_update(const float* __restrict__ aggpart, int nsplits,
                         const float* __restrict__ beta, const float* __restrict__ gamma,
                         const float* __restrict__ u_t, int* __restrict__ states,
                         _Float16* __restrict__ vh, float* __restrict__ out_t, int N) {
  int n = blockIdx.x * blockDim.x + threadIdx.x;
  if (n >= N) return;

  float agg = 0.0f;
  for (int k = 0; k < nsplits; ++k) agg += aggpart[(size_t)k * N + n];

  const int   s  = states[n];
  const float p0 = (s == 0) ? 1.0f : 0.0f;
  const float p1 = (s == 1) ? 1.0f : 0.0f;
  const float p2 = (s == 2) ? 1.0f : 0.0f;
  const float u  = u_t[n];

  float nc = beta[n] * (p0 * agg);
  float nr = gamma[n] * p1;
  float r0 = p0 - nc;
  float r1 = p1 + nc - nr;
  float r2 = p2 + nr;
  r0 = fmaxf(r0, 0.0f); r1 = fmaxf(r1, 0.0f); r2 = fmaxf(r2, 0.0f);
  float sum = (r0 + r1) + r2;
  float w0 = r0 / sum, w1 = r1 / sum, w2 = r2 / sum;

  bool condS = (p0 == 1.0f) && (u < w1 * 0.8f);
  bool condI = (p1 == 1.0f) && (u < w2);
  if (condS) { w0 = 0.0f; w1 = 1.0f; }
  if (condI) { w1 = 0.0f; w2 = 1.0f; }

  int next = 0; float m = w0;
  if (w1 > m) { m = w1; next = 1; }
  if (w2 > m) { next = 2; }

  out_t[3 * n + 0] = (next == 0) ? 1.0f : 0.0f;
  out_t[3 * n + 1] = (next == 1) ? 1.0f : 0.0f;
  out_t[3 * n + 2] = (next == 2) ? 1.0f : 0.0f;
  states[n] = next;
  vh[n] = (next == 1) ? (_Float16)1.0f : (_Float16)0.0f;
}

// ---------------------------------------------------------------------------
extern "C" void kernel_launch(void* const* d_in, const int* in_sizes, int n_in,
                              void* d_out, int out_size, void* d_ws, size_t ws_size,
                              hipStream_t stream) {
  const float* x     = (const float*)d_in[0];
  const float* H     = (const float*)d_in[1];
  const float* beta  = (const float*)d_in[2];
  const float* gamma = (const float*)d_in[3];
  const float* u     = (const float*)d_in[4];

  const int N     = in_sizes[2];                  // 20000
  const int STEPS = in_sizes[4] / N + 1;          // 16
  const int E     = (int)((long long)in_sizes[1] / STEPS / N);  // 1000

  const int KS1     = 125;                        // split-K for H@v
  const int KS2     = 8;                          // edge split for w^T H
  const int chunksN = N / 32;                     // 625 (N % 32 == 0)
  const int cpsN    = (chunksN + KS1 - 1) / KS1;  // 5
  const int chunksE = (E + 31) / 32;              // 32
  const int cpsE    = (chunksE + KS2 - 1) / KS2;  // 4
  const int ET      = (E + 15) / 16;              // 63 edge tiles
  const int NT      = N / 16;                     // 1250 node tiles

  // workspace carve (all offsets kept 32B-aligned)
  char* ws = (char*)d_ws;
  size_t off = 0;
  int*      states  = (int*)(ws + off);      off += ((size_t)N * 4 + 31) & ~(size_t)31;
  _Float16* vh      = (_Float16*)(ws + off); off += ((size_t)N * 2 + 31) & ~(size_t)31;
  _Float16* wh      = (_Float16*)(ws + off); off += ((size_t)WH_PAD * 2 + 31) & ~(size_t)31;
  float*    cpart   = (float*)(ws + off);    off += ((size_t)KS1 * WH_PAD * 4 + 31) & ~(size_t)31;
  float*    aggpart = (float*)(ws + off);    off += ((size_t)KS2 * N * 4 + 31) & ~(size_t)31;
  (void)ws_size; (void)n_in; (void)out_size;

  float* out = (float*)d_out;

  k_init<<<(N + 255) / 256, 256, 0, stream>>>(x, out, states, vh, N);

  for (int t = 1; t < STEPS; ++t) {
    const float* Hstep = H + (size_t)(t - 1) * (size_t)E * (size_t)N;
    k_edge<<<dim3(ET, KS1), 32, 0, stream>>>(Hstep, vh, cpart, N, chunksN, cpsN);
    k_tanh<<<1, 256, 0, stream>>>(cpart, KS1, wh, E);
    k_agg<<<dim3(NT, KS2), 32, 0, stream>>>(Hstep, wh, aggpart, N, chunksE, cpsE);
    k_update<<<(N + 255) / 256, 256, 0, stream>>>(aggpart, KS2, beta, gamma,
                                                  u + (size_t)(t - 1) * N,
                                                  states, vh,
                                                  out + (size_t)t * N * 3, N);
  }
}
